// PyTorchSSM_78915729096793
// MI455X (gfx1250) — matrix operand order repaired and checked by reading, not compiled, not run on hardware
//
#include <hip/hip_runtime.h>
#include <hip/hip_bf16.h>
#include <cstdint>
#include <cstddef>

// ---------------- problem constants (match reference) ----------------
#define D_MODEL   1024
#define D_STATE   128
#define D_CONV    4
#define HEADDIM   64
#define D_INNER   2048
#define NHEADS    32
#define CONV_DIM  2304          // D_INNER + 2*D_STATE
#define D_IN_PROJ 4384          // 2*D_INNER + 2*D_STATE + NHEADS
#define BATCH     2
#define SEQLEN    2048
#define NROWS     (BATCH * SEQLEN)   // 4096
#define EPS       1e-5f

typedef __bf16 bf16;
typedef __attribute__((ext_vector_type(16))) __bf16 v16bf;
typedef __attribute__((ext_vector_type(8)))  __bf16 v8bf;
typedef __attribute__((ext_vector_type(8)))  float  v8f;

// ---------------- fp32 -> bf16 (RNE) conversion ----------------
__global__ void f32_to_bf16_kernel(const float* __restrict__ in,
                                   bf16* __restrict__ out, int n) {
    int i = blockIdx.x * blockDim.x + threadIdx.x;
    if (i < n) out[i] = (bf16)in[i];
}

// ---------------- WMMA fragment loads -------------------------------
// A fragment: 16x32 bf16 tile of A[M,K] row-major (ld = K).
// Lanes 0-15: row M=lane, K = {0..7, 16..23}; lanes 16-31: K = {8..15, 24..31}.
__device__ __forceinline__ v16bf load_frag_a(const bf16* __restrict__ A,
                                             int ld, int row0, int k0, int lane) {
    int half = lane >> 4;
    int r    = lane & 15;
    const bf16* p = A + (size_t)(row0 + r) * ld + (k0 + half * 8);
    v8bf lo = *reinterpret_cast<const v8bf*>(p);        // K +0..7 (or +8..15)
    v8bf hi = *reinterpret_cast<const v8bf*>(p + 16);   // K +16..23 (or +24..31)
    return __builtin_shufflevector(lo, hi, 0,1,2,3,4,5,6,7,8,9,10,11,12,13,14,15);
}

// B fragment: 32x16 bf16 tile of W[N,K] row-major (ld = K) -> B = W^T slice.
// Lanes 0-15: col N=lane, K = {0..7, 16..23}; lanes 16-31: col N=lane-16, K = {8..15, 24..31}.
__device__ __forceinline__ v16bf load_frag_b(const bf16* __restrict__ W,
                                             int ld, int col0, int k0, int lane,
                                             int Nclamp) {
    int half = lane >> 4;
    int r    = lane & 15;
    int row  = col0 + r;
    if (row > Nclamp) row = Nclamp;                     // guard N tail (reads only)
    const bf16* p = W + (size_t)row * ld + (k0 + half * 8);
    v8bf lo = *reinterpret_cast<const v8bf*>(p);        // K +0..7 (or +8..15)
    v8bf hi = *reinterpret_cast<const v8bf*>(p + 16);   // K +16..23 (or +24..31)
    return __builtin_shufflevector(lo, hi, 0,1,2,3,4,5,6,7,8,9,10,11,12,13,14,15);
}

// ---------------- bf16 WMMA GEMM: C[M,N] = A[M,K] * W[N,K]^T ---------
// Block: 256 threads = 8 waves, waves tiled 4(M) x 2(N), each wave 32x32.
// Block tile = 128(M) x 64(N). M,K multiples of 128/32; N may be ragged.
__global__ void gemm_bf16_wmma(const bf16* __restrict__ A,
                               const bf16* __restrict__ W,
                               float* __restrict__ C,
                               int M, int N, int K,
                               int lda, int ldb, int ldc) {
    const int wave = threadIdx.x >> 5;
    const int lane = threadIdx.x & 31;
    const int wm   = wave & 3;
    const int wn   = wave >> 2;
    const int m0   = blockIdx.x * 128 + wm * 32;
    const int n0   = blockIdx.y * 64  + wn * 32;

    v8f acc00 = {}, acc01 = {}, acc10 = {}, acc11 = {};

    for (int k0 = 0; k0 < K; k0 += 32) {
        v16bf a0 = load_frag_a(A, lda, m0,      k0, lane);
        v16bf a1 = load_frag_a(A, lda, m0 + 16, k0, lane);
        v16bf b0 = load_frag_b(W, ldb, n0,      k0, lane, N - 1);
        v16bf b1 = load_frag_b(W, ldb, n0 + 16, k0, lane, N - 1);
        acc00 = __builtin_amdgcn_wmma_f32_16x16x32_bf16(false, a0, false, b0,
                                                        (short)0, acc00, false, false);
        acc01 = __builtin_amdgcn_wmma_f32_16x16x32_bf16(false, a0, false, b1,
                                                        (short)0, acc01, false, false);
        acc10 = __builtin_amdgcn_wmma_f32_16x16x32_bf16(false, a1, false, b0,
                                                        (short)0, acc10, false, false);
        acc11 = __builtin_amdgcn_wmma_f32_16x16x32_bf16(false, a1, false, b1,
                                                        (short)0, acc11, false, false);
    }

    // C/D layout: VGPR v -> M = v (+8 for lanes 16-31), N = lane&15.
    const int half = lane >> 4;
    const int r    = lane & 15;
    #pragma unroll
    for (int v = 0; v < 8; ++v) {
        int rr0 = m0 + v + half * 8;
        int rr1 = rr0 + 16;
        int cc0 = n0 + r;
        int cc1 = cc0 + 16;
        if (cc0 < N) {
            C[(size_t)rr0 * ldc + cc0] = acc00[v];
            C[(size_t)rr1 * ldc + cc0] = acc10[v];
        }
        if (cc1 < N) {
            C[(size_t)rr0 * ldc + cc1] = acc01[v];
            C[(size_t)rr1 * ldc + cc1] = acc11[v];
        }
    }
}

// ---------------- depthwise causal conv(k=4) + bias + SiLU -----------
__global__ void conv_silu_kernel(const float* __restrict__ zxbcdt,
                                 const float* __restrict__ conv_w,   // [CONV_DIM][1][4]
                                 const float* __restrict__ conv_b,
                                 float* __restrict__ xBC_act) {      // [NROWS][CONV_DIM]
    const int c = blockIdx.y * 256 + threadIdx.x;   // < CONV_DIM (9*256)
    const int m = blockIdx.x;                       // row = b*SEQLEN + l
    const int l = m & (SEQLEN - 1);
    const float* src = zxbcdt + (size_t)m * D_IN_PROJ + D_INNER + c;
    float acc = conv_b[c];
    #pragma unroll
    for (int j = 0; j < D_CONV; ++j) {
        int ll = l - (D_CONV - 1) + j;
        if (ll >= 0)
            acc += conv_w[c * D_CONV + j] * src[(ptrdiff_t)(j - (D_CONV - 1)) * D_IN_PROJ];
    }
    float sig = 1.f / (1.f + __expf(-acc));
    xBC_act[(size_t)m * CONV_DIM + c] = acc * sig;
}

// ---------------- selective scan: one block per (b,h) ----------------
// 256 threads: p = tid & 63 (HEADDIM), g = tid >> 6 (state group of 32).
// Each thread keeps 32 state elements in registers for all 2048 steps.
// Global -> LDS staging of x_t/B_t/C_t/dt_t is done with asynchronous
// copies (ASYNCcnt) into a 3-deep ring buffer, prefetching 2 steps ahead.
#define SCAN_ROW 324   // 320 staged floats + dt at [320]; 324*4B = 16B multiple

__global__ void ssm_scan_kernel(const float* __restrict__ zxbcdt,
                                const float* __restrict__ xBC_act,
                                const float* __restrict__ dt_bias,
                                const float* __restrict__ A_log,
                                const float* __restrict__ D_param,
                                float* __restrict__ y) {             // [NROWS][D_INNER]
    const int bh  = blockIdx.x;          // 0..63
    const int b   = bh >> 5;
    const int h   = bh & 31;
    const int tid = threadIdx.x;
    const int p   = tid & 63;
    const int g   = tid >> 6;            // 0..3

    const float Acoef = -__expf(A_log[h]);
    const float Dcoef = D_param[h];
    const float dtb   = dt_bias[h];

    float s[32];
    #pragma unroll
    for (int i = 0; i < 32; ++i) s[i] = 0.f;

    __shared__ alignas(16) float sh[3][SCAN_ROW]; // [0..63]=x, [64..191]=B, [192..319]=C, [320]=dt
    __shared__ float shpart[256];

    // Issue async prefetch of timestep t into ring buffer `buf`.
    // Waves 0-2: one b128 per stage (threads 0..79); wave 3: one b32 (thread 96).
    // => at most 1 async instruction per wave per stage (uniform wait count).
    auto stage = [&](int t, int buf) {
        const size_t row = (size_t)b * SEQLEN + t;
        if (tid < 80) {
            const int j = tid * 4;
            const float* xr = xBC_act + row * CONV_DIM;
            const float* src;
            if (j < 64)       src = xr + h * HEADDIM + j;
            else if (j < 192) src = xr + D_INNER + (j - 64);
            else              src = xr + D_INNER + D_STATE + (j - 192);
            uint32_t lds = (uint32_t)(uintptr_t)(void*)&sh[buf][j];
            asm volatile("global_load_async_to_lds_b128 %0, %1, off"
                         :: "v"(lds), "v"((uint64_t)(uintptr_t)src) : "memory");
        } else if (tid == 96) {
            const float* src = zxbcdt + row * D_IN_PROJ + (D_INNER + CONV_DIM) + h;
            uint32_t lds = (uint32_t)(uintptr_t)(void*)&sh[buf][320];
            asm volatile("global_load_async_to_lds_b32 %0, %1, off"
                         :: "v"(lds), "v"((uint64_t)(uintptr_t)src) : "memory");
        }
    };

    stage(0, 0);
    stage(1, 1);

    for (int t = 0; t < SEQLEN; ++t) {
        const int tn = (t + 2 < SEQLEN) ? (t + 2) : (SEQLEN - 1);
        stage(tn, (t + 2) % 3);     // overwrites buffer last read at step t-1

        // Two newer stages may remain in flight; async loads complete in
        // order, so asynccnt<=2 means stage t's copies have landed in LDS.
        asm volatile("s_wait_asynccnt 0x2" ::: "memory");
        __syncthreads();            // publish all waves' staged data

        const float* S = sh[t % 3];
        float dtr   = S[320] + dtb;
        float dts   = (dtr > 20.f) ? dtr : log1pf(__expf(dtr));
        float decay = __expf(Acoef * dts);

        float xdt  = S[p] * dts;
        float part = 0.f;
        const float* Bp = S + 64  + g * 32;
        const float* Cp = S + 192 + g * 32;
        #pragma unroll
        for (int i = 0; i < 32; ++i) {
            s[i] = s[i] * decay + xdt * Bp[i];
            part += s[i] * Cp[i];
        }
        shpart[tid] = part;
        __syncthreads();

        if (tid < 64) {
            const size_t row = (size_t)b * SEQLEN + t;
            float yv = shpart[tid] + shpart[tid + 64] + shpart[tid + 128] +
                       shpart[tid + 192] + Dcoef * S[tid];
            y[row * D_INNER + h * HEADDIM + tid] = yv;
        }
        __syncthreads();   // protect ring buffer + shpart before next step
    }
}

// ---------------- gate with SiLU(z), RMSNorm, emit bf16 --------------
__global__ void gate_norm_kernel(const float* __restrict__ y,
                                 const float* __restrict__ zxbcdt,
                                 const float* __restrict__ norm_w,
                                 bf16* __restrict__ ybf) {           // [NROWS][D_INNER]
    const int m   = blockIdx.x;
    const int tid = threadIdx.x;     // 256 threads, 8 elems each
    __shared__ float red[256];

    const float* yr = y + (size_t)m * D_INNER;
    const float* zr = zxbcdt + (size_t)m * D_IN_PROJ;   // z = first D_INNER cols

    float vals[8];
    float ss = 0.f;
    #pragma unroll
    for (int i = 0; i < 8; ++i) {
        int e = tid + i * 256;
        float z = zr[e];
        float v = yr[e] * (z / (1.f + __expf(-z)));
        vals[i] = v;
        ss += v * v;
    }
    red[tid] = ss;
    __syncthreads();
    for (int s2 = 128; s2 > 0; s2 >>= 1) {
        if (tid < s2) red[tid] += red[tid + s2];
        __syncthreads();
    }
    float inv = rsqrtf(red[0] * (1.f / (float)D_INNER) + EPS);
    #pragma unroll
    for (int i = 0; i < 8; ++i) {
        int e = tid + i * 256;
        ybf[(size_t)m * D_INNER + e] = (bf16)(vals[i] * inv * norm_w[e]);
    }
}

// ---------------- launcher ------------------------------------------
extern "C" void kernel_launch(void* const* d_in, const int* in_sizes, int n_in,
                              void* d_out, int out_size, void* d_ws, size_t ws_size,
                              hipStream_t stream) {
    const float* x          = (const float*)d_in[0];  // [2,2048,1024]
    const float* in_proj_w  = (const float*)d_in[1];  // [4384,1024]
    const float* conv_w     = (const float*)d_in[2];  // [2304,1,4]
    const float* conv_b     = (const float*)d_in[3];  // [2304]
    const float* norm_w     = (const float*)d_in[4];  // [2048]
    const float* out_proj_w = (const float*)d_in[5];  // [1024,2048]
    const float* dt_bias    = (const float*)d_in[6];  // [32]
    const float* A_log      = (const float*)d_in[7];  // [32]
    const float* D_param    = (const float*)d_in[8];  // [32]
    float* out = (float*)d_out;                       // [2,2048,1024] fp32

    // ---- workspace layout (bytes) ----
    char* ws = (char*)d_ws;
    size_t off = 0;
    float* zxbcdt  = (float*)(ws + off); off += (size_t)NROWS * D_IN_PROJ * 4; // 71.8 MB
    float* xBC_act = (float*)(ws + off); off += (size_t)NROWS * CONV_DIM * 4;  // 37.7 MB
    float* ybuf    = (float*)(ws + off); off += (size_t)NROWS * D_INNER * 4;   // 33.6 MB
    bf16*  x_bf    = (bf16*) (ws + off); off += (size_t)NROWS * D_MODEL * 2;   //  8.4 MB
    bf16*  w1_bf   = (bf16*) (ws + off); off += (size_t)D_IN_PROJ * D_MODEL * 2; // 9 MB
    bf16*  w2_bf   = (bf16*) (ws + off); off += (size_t)D_MODEL * D_INNER * 2; //  4.2 MB
    bf16*  y_bf    = (bf16*) (ws + off); off += (size_t)NROWS * D_INNER * 2;   // 16.8 MB

    // 1) bf16 staging of GEMM operands
    {
        int n1 = NROWS * D_MODEL;
        int n2 = D_IN_PROJ * D_MODEL;
        int n3 = D_MODEL * D_INNER;
        f32_to_bf16_kernel<<<(n1 + 255) / 256, 256, 0, stream>>>(x, x_bf, n1);
        f32_to_bf16_kernel<<<(n2 + 255) / 256, 256, 0, stream>>>(in_proj_w, w1_bf, n2);
        f32_to_bf16_kernel<<<(n3 + 255) / 256, 256, 0, stream>>>(out_proj_w, w2_bf, n3);
    }

    // 2) in_proj GEMM: zxbcdt[4096,4384] = x[4096,1024] * W1[4384,1024]^T
    {
        dim3 grid(NROWS / 128, (D_IN_PROJ + 63) / 64);
        gemm_bf16_wmma<<<grid, 256, 0, stream>>>(x_bf, w1_bf, zxbcdt,
                                                 NROWS, D_IN_PROJ, D_MODEL,
                                                 D_MODEL, D_MODEL, D_IN_PROJ);
    }

    // 3) depthwise conv + bias + SiLU
    {
        dim3 grid(NROWS, CONV_DIM / 256);
        conv_silu_kernel<<<grid, 256, 0, stream>>>(zxbcdt, conv_w, conv_b, xBC_act);
    }

    // 4) selective scan (64 independent (b,h) blocks, async-prefetched)
    ssm_scan_kernel<<<BATCH * NHEADS, 256, 0, stream>>>(zxbcdt, xBC_act,
                                                        dt_bias, A_log, D_param, ybuf);

    // 5) gate + RMSNorm -> bf16
    gate_norm_kernel<<<NROWS, 256, 0, stream>>>(ybuf, zxbcdt, norm_w, y_bf);

    // 6) out_proj GEMM: out[4096,1024] = y[4096,2048] * W2[1024,2048]^T
    {
        dim3 grid(NROWS / 128, D_MODEL / 64);
        gemm_bf16_wmma<<<grid, 256, 0, stream>>>(y_bf, w2_bf, out,
                                                 NROWS, D_MODEL, D_INNER,
                                                 D_INNER, D_INNER, D_MODEL);
    }
    (void)in_sizes; (void)n_in; (void)out_size; (void)ws_size;
}
